// MMGCNModel_35107062678319
// MI455X (gfx1250) — matrix operand order repaired and checked by reading, not compiled, not run on hardware
//
#include <hip/hip_runtime.h>

#define U_CNT 20000
#define I_CNT 20000
#define N_CNT 40000
#define K_DIM 64
#define D_DIM 256
#define E_CNT 640000
#define B_CNT 4096

typedef __attribute__((ext_vector_type(16))) __bf16 v16bf;
typedef __attribute__((ext_vector_type(4)))  __bf16 v4bf;
typedef __attribute__((ext_vector_type(8)))  float  v8f;

__device__ __forceinline__ float lrelu(float x) { return x > 0.0f ? x : 0.01f * x; }

// ------------------------------------------------------------------
// WMMA GEMM: C[M,Nc] = A[M,Kd] @ B[Kd,Nc] (+ bias[Nc] if non-null)
// Block = 8 waves, 128x64 tile, BK=32. fp32 -> bf16 in LDS,
// fp32 accumulate via v_wmma_f32_16x16x32_bf16 (4 per wave per K-step).
// Software-pipelined; grid is (col-tiles, row-tiles) so blocks sharing
// an A row-strip run back-to-back and A re-reads hit the 192MB L2.
// ------------------------------------------------------------------
__global__ __launch_bounds__(256) void gemm_wmma(
    const float* __restrict__ A, const float* __restrict__ B,
    const float* __restrict__ bias, float* __restrict__ C,
    int M, int Kd, int Nc)
{
  __shared__ __bf16 As[128][32];  // [row][k]
  __shared__ __bf16 Bs[64][32];   // transposed: [col][k]
  const int tid  = threadIdx.x;
  const int lane = tid & 31;
  const int wave = tid >> 5;      // wave's 16-row strip (0..7)
  const int bn = blockIdx.x * 64;   // fast dim: column tiles
  const int bm = blockIdx.y * 128;  // slow dim: row tiles (L2 reuse of A)
  const int h  = lane >> 4;       // lane half (K-group select)
  const int ml = lane & 15;       // row (A) / col (B,C)

  // A staging: 4 x float4 per thread (rows ar0+32i, k = ak..ak+3)
  const int ar0 = tid >> 3;          // 0..31
  const int ak  = (tid & 7) << 2;    // 0..28 step 4
  // B staging: 2 groups of 4 coalesced b32 along K for one column
  const int bcol = tid & 63;
  const int bk0  = (tid >> 6) << 2;  // 0,4,8,12

  v8f acc[4] = {{}, {}, {}, {}};
  float4 ra[4];
  float  rb[2][4];

  auto load_tile = [&](int k0) {
    #pragma unroll
    for (int i = 0; i < 4; ++i) {
      int row = bm + ar0 + i * 32;
      row = row < M ? row : M - 1;   // clamp: dup loads, never stored
      ra[i] = *reinterpret_cast<const float4*>(&A[(long)row * Kd + (k0 + ak)]);
    }
    #pragma unroll
    for (int g = 0; g < 2; ++g) {
      int kk = k0 + bk0 + g * 16;
      #pragma unroll
      for (int j = 0; j < 4; ++j)
        rb[g][j] = B[(long)(kk + j) * Nc + (bn + bcol)];
    }
  };

  auto store_tile = [&]() {
    #pragma unroll
    for (int i = 0; i < 4; ++i) {
      v4bf v;
      v[0] = (__bf16)ra[i].x; v[1] = (__bf16)ra[i].y;
      v[2] = (__bf16)ra[i].z; v[3] = (__bf16)ra[i].w;
      *reinterpret_cast<v4bf*>(&As[ar0 + i * 32][ak]) = v;
    }
    #pragma unroll
    for (int g = 0; g < 2; ++g) {
      v4bf v;
      v[0] = (__bf16)rb[g][0]; v[1] = (__bf16)rb[g][1];
      v[2] = (__bf16)rb[g][2]; v[3] = (__bf16)rb[g][3];
      *reinterpret_cast<v4bf*>(&Bs[bcol][bk0 + g * 16]) = v;
    }
  };

  load_tile(0);
  for (int k0 = 0; k0 < Kd; k0 += 32) {
    store_tile();
    __syncthreads();
    if (k0 + 32 < Kd) load_tile(k0 + 32);   // prefetch under compute

    // fragments: contiguous k = 16*h + i for both A and B (wave32)
    v16bf fa;
    #pragma unroll
    for (int i = 0; i < 16; ++i) {
      int v = i >> 1, p = i & 1;
      int ka = (h << 4) + (v << 1) + p;
      fa[i] = As[wave * 16 + ml][ka];
    }
    #pragma unroll
    for (int t = 0; t < 4; ++t) {
      v16bf fb;
      #pragma unroll
      for (int i = 0; i < 16; ++i) {
        int v = i >> 1, p = i & 1;
        int kb = (h << 4) + (v << 1) + p;
        fb[i] = Bs[t * 16 + ml][kb];
      }
      acc[t] = __builtin_amdgcn_wmma_f32_16x16x32_bf16(false, fa, false, fb,
                                                       (short)0, acc[t], false, false);
    }
    __syncthreads();
  }

  // C/D layout: VGPR v -> row = strip + half*8 + v, col = ml
  #pragma unroll
  for (int t = 0; t < 4; ++t) {
    int col = bn + t * 16 + ml;
    float bv = bias ? bias[col] : 0.0f;
    #pragma unroll
    for (int v = 0; v < 8; ++v) {
      int row = bm + wave * 16 + h * 8 + v;
      if (row < M)
        C[(long)row * Nc + col] = acc[t][v] + bv;
    }
  }
}

// ---------------- graph / elementwise kernels ----------------
__global__ void k_count(const int* __restrict__ dst, float* cnt, int E)
{
  int e = blockIdx.x * blockDim.x + threadIdx.x;
  if (e < E) atomicAdd(&cnt[dst[e]], 1.0f);
}

// one block per edge; each thread gathers a float4 and does 4 atomics
// (y and s both fit in the 192MB L2, so this phase is L2-resident)
__global__ void k_scatter4(const float* __restrict__ y, const int* __restrict__ src,
                           const int* __restrict__ dst, float* s, int dq /* d/4 */)
{
  int e = blockIdx.x;
  int t = threadIdx.x;
  if (t >= dq) return;
  int se = src[e], de = dst[e];
  float4 v = reinterpret_cast<const float4*>(y + (long)se * dq * 4)[t];
  float* sr = s + (long)de * dq * 4 + t * 4;
  atomicAdd(sr + 0, v.x);
  atomicAdd(sr + 1, v.y);
  atomicAdd(sr + 2, v.z);
  atomicAdd(sr + 3, v.w);
}

__global__ void k_l2norm(const float* __restrict__ in, float* out, int D)
{
  __shared__ float red[8];
  long r = blockIdx.x;
  float ss = 0.0f;
  for (int c = threadIdx.x; c < D; c += blockDim.x) {
    float v = in[r * D + c];
    ss += v * v;
  }
  for (int off = 16; off; off >>= 1) ss += __shfl_down(ss, off, 32);
  int lane = threadIdx.x & 31, w = threadIdx.x >> 5;
  if (lane == 0) red[w] = ss;
  __syncthreads();
  if (w == 0) {
    float t = (lane < 8) ? red[lane] : 0.0f;
    for (int off = 4; off; off >>= 1) t += __shfl_down(t, off, 32);
    if (lane == 0) red[0] = t;
  }
  __syncthreads();
  float inv = 1.0f / fmaxf(sqrtf(red[0]), 1e-12f);
  for (int c = threadIdx.x; c < D; c += blockDim.x)
    out[r * D + c] = in[r * D + c] * inv;
}

// s = leaky(s / max(cnt,1)) in place, float4; shift = log2(d)
__global__ void k_mean_leaky4(float* s, const float* __restrict__ cnt,
                              long total4, int shift)
{
  long i = (long)blockIdx.x * blockDim.x + threadIdx.x;
  if (i >= total4) return;
  long b = i << 2;                       // 4 elems share a row (d % 4 == 0)
  float inv = 1.0f / fmaxf(cnt[b >> shift], 1.0f);
  float4 v = reinterpret_cast<float4*>(s)[i];
  v.x = lrelu(v.x * inv); v.y = lrelu(v.y * inv);
  v.z = lrelu(v.z * inv); v.w = lrelu(v.w * inv);
  reinterpret_cast<float4*>(s)[i] = v;
}

// x_out = leaky(hg + leaky(xl) + ego), float4 over [N,64]
__global__ void k_combine4(const float* __restrict__ hg, const float* __restrict__ xl,
                           const float* __restrict__ Gu, const float* __restrict__ Gi,
                           float* xout, int Nrows, int Urows)
{
  long i = (long)blockIdx.x * blockDim.x + threadIdx.x;   // float4 index
  if (i >= (long)Nrows * 16) return;
  long b = i << 2;
  int r = (int)(b >> 6);
  int c = (int)(b & 63);
  const float* ego = (r < Urows) ? (Gu + (long)r * 64 + c)
                                 : (Gi + (long)(r - Urows) * 64 + c);
  float4 e4 = *reinterpret_cast<const float4*>(ego);
  float4 g4 = reinterpret_cast<const float4*>(hg)[i];
  float4 l4 = reinterpret_cast<const float4*>(xl)[i];
  float4 o;
  o.x = lrelu(g4.x + lrelu(l4.x) + e4.x);
  o.y = lrelu(g4.y + lrelu(l4.y) + e4.y);
  o.z = lrelu(g4.z + lrelu(l4.z) + e4.z);
  o.w = lrelu(g4.w + lrelu(l4.w) + e4.w);
  reinterpret_cast<float4*>(xout)[i] = o;
}

__global__ void k_add4(float* acc, const float* __restrict__ x, long n4)
{
  long i = (long)blockIdx.x * blockDim.x + threadIdx.x;
  if (i >= n4) return;
  float4 a = reinterpret_cast<float4*>(acc)[i];
  float4 v = reinterpret_cast<const float4*>(x)[i];
  a.x += v.x; a.y += v.y; a.z += v.z; a.w += v.w;
  reinterpret_cast<float4*>(acc)[i] = a;
}

// out[b] = 0.25 * dot(acc[users[b]], acc[U+items[b]])  (0.5 modality mean twice)
__global__ void k_dot(const float* __restrict__ acc, const int* __restrict__ users,
                      const int* __restrict__ items, float* out, int Bn, int Urows)
{
  int b = blockIdx.x * blockDim.x + threadIdx.x;
  if (b >= Bn) return;
  const float4* pu = reinterpret_cast<const float4*>(acc + (long)users[b] * 64);
  const float4* pi = reinterpret_cast<const float4*>(acc + (long)(Urows + items[b]) * 64);
  float s = 0.0f;
  #pragma unroll
  for (int k = 0; k < 16; ++k) {
    float4 a = pu[k], c = pi[k];
    s += a.x * c.x + a.y * c.y + a.z * c.z + a.w * c.w;
  }
  out[b] = 0.25f * s;
}

// ------------------------------------------------------------------
extern "C" void kernel_launch(void* const* d_in, const int* in_sizes, int n_in,
                              void* d_out, int out_size, void* d_ws, size_t ws_size,
                              hipStream_t stream)
{
  (void)in_sizes; (void)n_in; (void)out_size; (void)ws_size;

  // setup_inputs() dict-insertion order, params flattened depth-first
  const float* Gu = (const float*)d_in[0];
  const float* Gi = (const float*)d_in[1];
  const int moff[2] = {2, 15};                       // per-modality leaf offsets
  const float* Fm[2] = {(const float*)d_in[28], (const float*)d_in[29]};
  const int fdim[2] = {2048, 768};
  const int* edge  = (const int*)d_in[30];
  const int* srcI  = edge;                            // edge_index[0]
  const int* dstI  = edge + E_CNT;                    // edge_index[1]
  const int* users = (const int*)d_in[31];
  const int* items = (const int*)d_in[32];
  float* out = (float*)d_out;

  // workspace carve (256B aligned): ~144 MB total
  char* w = (char*)d_ws;
  auto carve = [&](size_t bytes) -> void* {
    void* p = (void*)w;
    w += (bytes + 255) & ~(size_t)255;
    return p;
  };
  float* cnt  = (float*)carve((size_t)N_CNT * 4);
  float* x256 = (float*)carve((size_t)N_CNT * D_DIM * 4);
  float* y256 = (float*)carve((size_t)N_CNT * D_DIM * 4);  // y, then xl/hg reuse
  float* s256 = (float*)carve((size_t)N_CNT * D_DIM * 4);  // scatter accum -> h
  float* x64  = (float*)carve((size_t)N_CNT * K_DIM * 4);
  float* acc  = (float*)carve((size_t)N_CNT * K_DIM * 4);
  float* xl = y256;                         // y is dead after scatter
  float* hg = y256 + (size_t)N_CNT * K_DIM;

  hipMemsetAsync(cnt, 0, (size_t)N_CNT * 4, stream);
  hipMemsetAsync(acc, 0, (size_t)N_CNT * K_DIM * 4, stream);
  k_count<<<(E_CNT + 255) / 256, 256, 0, stream>>>(dstI, cnt, E_CNT);

  for (int m = 0; m < 2; ++m) {
    const float* Gum = (const float*)d_in[moff[m] + 0];
    const float* pW  = (const float*)d_in[moff[m] + 1];
    const float* pb  = (const float*)d_in[moff[m] + 2];
    const float* propW[2] = {(const float*)d_in[moff[m] + 3], (const float*)d_in[moff[m] + 4]};
    const float* linW[2]  = {(const float*)d_in[moff[m] + 5], (const float*)d_in[moff[m] + 6]};
    const float* linb[2]  = {(const float*)d_in[moff[m] + 7], (const float*)d_in[moff[m] + 8]};
    const float* gW[2]    = {(const float*)d_in[moff[m] + 9], (const float*)d_in[moff[m] + 10]};
    const float* gb[2]    = {(const float*)d_in[moff[m] + 11], (const float*)d_in[moff[m] + 12]};

    // feats = F @ proj_W + proj_b  -> rows [U, N) of x256
    dim3 gproj(D_DIM / 64, (I_CNT + 127) / 128);
    gemm_wmma<<<gproj, 256, 0, stream>>>(Fm[m], pW, pb,
                                         x256 + (size_t)U_CNT * D_DIM,
                                         I_CNT, fdim[m], D_DIM);
    hipMemcpyAsync(x256, Gum, (size_t)U_CNT * D_DIM * 4,
                   hipMemcpyDeviceToDevice, stream);
    k_l2norm<<<N_CNT, 256, 0, stream>>>(x256, x256, D_DIM);

    const float* x = x256;
    int d = D_DIM;
    for (int l = 0; l < 2; ++l) {
      int shift = (d == 256) ? 8 : 6;
      // y = x @ prop_W  (compute-then-gather: N rows instead of E)
      dim3 gy(d / 64, (N_CNT + 127) / 128);
      gemm_wmma<<<gy, 256, 0, stream>>>(x, propW[l], nullptr, y256, N_CNT, d, d);
      hipMemsetAsync(s256, 0, (size_t)N_CNT * d * 4, stream);
      k_scatter4<<<E_CNT, d / 4, 0, stream>>>(y256, srcI, dstI, s256, d / 4);
      long tot4 = (long)N_CNT * d / 4;
      k_mean_leaky4<<<(int)((tot4 + 255) / 256), 256, 0, stream>>>(s256, cnt, tot4, shift);
      // xl = x @ lin_W + lin_b ; hg = h @ g_W + g_b
      dim3 gk(1, (N_CNT + 127) / 128);
      gemm_wmma<<<gk, 256, 0, stream>>>(x, linW[l], linb[l], xl, N_CNT, d, K_DIM);
      gemm_wmma<<<gk, 256, 0, stream>>>(s256, gW[l], gb[l], hg, N_CNT, d, K_DIM);
      long t16 = (long)N_CNT * 16;
      k_combine4<<<(int)((t16 + 255) / 256), 256, 0, stream>>>(hg, xl, Gu, Gi, x64,
                                                               N_CNT, U_CNT);
      x = x64; d = K_DIM;
    }
    k_add4<<<(int)(((long)N_CNT * 16 + 255) / 256), 256, 0, stream>>>(
        acc, x64, (long)N_CNT * 16);
  }
  k_dot<<<(B_CNT + 255) / 256, 256, 0, stream>>>(acc, users, items, out, B_CNT, U_CNT);
}
